// UnifiedResampler_75625784148332
// MI455X (gfx1250) — matrix operand order repaired and checked by reading, not compiled, not run on hardware
//
#include <hip/hip_runtime.h>

typedef __attribute__((ext_vector_type(16))) __bf16 v16bf;
typedef __attribute__((ext_vector_type(8)))  float  v8f;

#define LN_EPS 1e-5f

// ---------------- fp32 -> bf16 (weights), vectorized x4, grid-stride ----------------
__global__ __launch_bounds__(256) void cvt_f32_bf16(const float* __restrict__ src,
                                                    __bf16* __restrict__ dst, int n4) {
    int i = blockIdx.x * blockDim.x + threadIdx.x;
    int stride = gridDim.x * blockDim.x;
    for (; i < n4; i += stride) {
        float4 f = ((const float4*)src)[i];
        union { __bf16 h[4]; uint2 u; } o;
        o.h[0] = (__bf16)f.x; o.h[1] = (__bf16)f.y;
        o.h[2] = (__bf16)f.z; o.h[3] = (__bf16)f.w;
        ((uint2*)dst)[i] = o.u;
    }
}

// ---------------- LayerNorm(x_q) -> bf16, one block per row of 4096 ----------------
__global__ __launch_bounds__(256) void ln_q_kernel(const float* __restrict__ x,
                                                   const float* __restrict__ w,
                                                   const float* __restrict__ b,
                                                   __bf16* __restrict__ out) {
    const int row = blockIdx.x;
    const int tid = threadIdx.x;
    const float* xr = x + (size_t)row * 4096;

    float4 v[4];
    float s = 0.f, q = 0.f;
#pragma unroll
    for (int i = 0; i < 4; ++i) {
        v[i] = *(const float4*)(xr + tid * 16 + i * 4);
        s += v[i].x + v[i].y + v[i].z + v[i].w;
        q += v[i].x * v[i].x + v[i].y * v[i].y + v[i].z * v[i].z + v[i].w * v[i].w;
    }
    __shared__ float ss[256], sq[256];
    ss[tid] = s; sq[tid] = q; __syncthreads();
    for (int st = 128; st > 0; st >>= 1) {
        if (tid < st) { ss[tid] += ss[tid + st]; sq[tid] += sq[tid + st]; }
        __syncthreads();
    }
    const float mean = ss[0] * (1.f / 4096.f);
    const float var  = sq[0] * (1.f / 4096.f) - mean * mean;
    const float inv  = rsqrtf(var + LN_EPS);

    union { __bf16 h[16]; int4 q4[2]; } o;
#pragma unroll
    for (int i = 0; i < 4; ++i) {
        const float* f = (const float*)&v[i];
#pragma unroll
        for (int j = 0; j < 4; ++j) {
            const int col = tid * 16 + i * 4 + j;
            o.h[i * 4 + j] = (__bf16)((f[j] - mean) * inv * w[col] + b[col]);
        }
    }
    int4* op = (int4*)(out + (size_t)row * 4096 + tid * 16);
    op[0] = o.q4[0]; op[1] = o.q4[1];
}

// ------- LayerNorm(x_kv) with window permutation folded in; emits k-LN and v-LN rows -------
__global__ __launch_bounds__(256) void ln_kv_kernel(const float* __restrict__ x,
                                                    const float* __restrict__ kw,
                                                    const float* __restrict__ kb,
                                                    const float* __restrict__ vw,
                                                    const float* __restrict__ vb,
                                                    __bf16* __restrict__ outK,
                                                    __bf16* __restrict__ outV) {
    const int d   = blockIdx.x;      // dest row 0..18431 in (b, p, s) order
    const int tid = threadIdx.x;
    const int bb  = d / 2304;
    const int rem = d % 2304;
    const int p   = rem >> 2;        // 0..575
    const int s   = rem & 3;         // 0..3
    const int ph = p / 24, pw = p % 24;
    const int i2 = s >> 1, j2 = s & 1;
    const int src = bb * 2304 + (ph * 2 + i2) * 48 + (pw * 2 + j2);

    const float* xr = x + (size_t)src * 1024;
    float4 v = *(const float4*)(xr + tid * 4);
    float sum = v.x + v.y + v.z + v.w;
    float ssq = v.x * v.x + v.y * v.y + v.z * v.z + v.w * v.w;

    __shared__ float ss[256], sq[256];
    ss[tid] = sum; sq[tid] = ssq; __syncthreads();
    for (int st = 128; st > 0; st >>= 1) {
        if (tid < st) { ss[tid] += ss[tid + st]; sq[tid] += sq[tid + st]; }
        __syncthreads();
    }
    const float mean = ss[0] * (1.f / 1024.f);
    const float var  = sq[0] * (1.f / 1024.f) - mean * mean;
    const float inv  = rsqrtf(var + LN_EPS);

    const float* f = (const float*)&v;
    union { __bf16 h[4]; uint2 u; } ok, ov;
#pragma unroll
    for (int j = 0; j < 4; ++j) {
        const int col = tid * 4 + j;
        const float n = (f[j] - mean) * inv;
        ok.h[j] = (__bf16)(n * kw[col] + kb[col]);
        ov.h[j] = (__bf16)(n * vw[col] + vb[col]);
    }
    *(uint2*)(outK + (size_t)d * 1024 + tid * 4) = ok.u;
    *(uint2*)(outV + (size_t)d * 1024 + tid * 4) = ov.u;
}

// ---------------- bf16 WMMA GEMM: C[M,N] = A[M,K] @ W[K,N] + bias ----------------
// 128x256 block tile, BK=32, 8 waves (2x4), each wave 64x64 = 16 WMMA accumulators.
// A fragments reused across 4 N-tiles, B fragments across 4 M-tiles.
template <bool OUT_BF16>
__global__ __launch_bounds__(256)
void gemm_bf16_kernel(const __bf16* __restrict__ A, const __bf16* __restrict__ W,
                      const float* __restrict__ bias,
                      float* __restrict__ Cf, __bf16* __restrict__ Cb,
                      int M, int N, int K) {
    __shared__ __align__(16) __bf16 sA[2][128][40];   // row-major [m][k], padded (20 KB)
    __shared__ __align__(16) __bf16 sB[2][256][40];   // transposed  [n][k], padded (40 KB)

    const int tid  = threadIdx.x;
    const int lane = tid & 31;
    const int wave = tid >> 5;
    const int wm = (wave >> 2) * 64;   // wave M offset (0 / 64)
    const int wn = (wave & 3) * 64;    // wave N offset (0/64/128/192)
    const int lr = lane & 15;
    const int lh = lane >> 4;

    const int mb0 = blockIdx.y * 128;
    const int nb0 = blockIdx.x * 256;

    // global-load mapping: A -> 16 bf16 per thread (row t/2, cols (t&1)*16..)
    //                      B -> 4 k-rows x 8 n per thread, stored transposed
    const int arow = tid >> 1;
    const int acol = (tid & 1) * 16;
    const int bkk  = (tid >> 5) * 4;   // 0,4,...,28
    const int bn0  = (tid & 31) * 8;   // 0..248

    const __bf16* Aptr = A + (size_t)(mb0 + arow) * K + acol;
    const __bf16* Bptr = W + (size_t)bkk * N + nb0 + bn0;

    int4 ra0, ra1, rb[4];

    auto loadg = [&](int kt) {
        const __bf16* ap = Aptr + kt * 32;
        ra0 = *(const int4*)(ap);
        ra1 = *(const int4*)(ap + 8);
        const __bf16* bp = Bptr + (size_t)kt * 32 * N;
#pragma unroll
        for (int r = 0; r < 4; ++r) rb[r] = *(const int4*)(bp + (size_t)r * N);
        __builtin_prefetch((const void*)(ap + 32), 0, 0);             // next A tile
        __builtin_prefetch((const void*)(bp + (size_t)32 * N), 0, 0); // next B tile
    };
    auto stage = [&](int buf) {
        *(int4*)&sA[buf][arow][acol]     = ra0;
        *(int4*)&sA[buf][arow][acol + 8] = ra1;
        const __bf16* r0 = (const __bf16*)&rb[0];
        const __bf16* r1 = (const __bf16*)&rb[1];
        const __bf16* r2 = (const __bf16*)&rb[2];
        const __bf16* r3 = (const __bf16*)&rb[3];
#pragma unroll
        for (int e = 0; e < 8; ++e) {
            union { __bf16 h[4]; uint2 u; } pk;
            pk.h[0] = r0[e]; pk.h[1] = r1[e]; pk.h[2] = r2[e]; pk.h[3] = r3[e];
            *(uint2*)&sB[buf][bn0 + e][bkk] = pk.u;   // 8B-aligned (bkk % 4 == 0)
        }
    };

    v8f acc[4][4];
    const v8f vzero = {};
#pragma unroll
    for (int mi = 0; mi < 4; ++mi)
#pragma unroll
        for (int nj = 0; nj < 4; ++nj) acc[mi][nj] = vzero;

    const int nk = K >> 5;
    loadg(0);
    stage(0);
    __syncthreads();

    for (int kt = 0; kt < nk; ++kt) {
        const int cur = kt & 1;
        if (kt + 1 < nk) loadg(kt + 1);

        union FA { v16bf v; int4 q[2]; } fa[4];
#pragma unroll
        for (int mi = 0; mi < 4; ++mi) {
            // A lane layout: lanes 0-15: K[0:8]+K[16:24]; lanes 16-31: K[8:16]+K[24:32]
            const __bf16* ar = &sA[cur][wm + mi * 16 + lr][lh * 8];
            fa[mi].q[0] = *(const int4*)(ar);
            fa[mi].q[1] = *(const int4*)(ar + 16);
        }
#pragma unroll
        for (int nj = 0; nj < 4; ++nj) {
            // B lane layout: lanes 0-15: K[0:8]+K[16:24]; lanes 16-31: K[8:16]+K[24:32]
            union FB { v16bf v; int4 q[2]; } fb;
            const __bf16* bc = &sB[cur][wn + nj * 16 + lr][lh * 8];
            fb.q[0] = *(const int4*)(bc);
            fb.q[1] = *(const int4*)(bc + 16);
#pragma unroll
            for (int mi = 0; mi < 4; ++mi)
                acc[mi][nj] = __builtin_amdgcn_wmma_f32_16x16x32_bf16(
                    false, fa[mi].v, false, fb.v, (short)0, acc[mi][nj],
                    false, false);
        }

        if (kt + 1 < nk) stage(cur ^ 1);
        __syncthreads();
    }

    // epilogue: C/D layout -> VGPR r: (M = lh*8 + r, N = lr)
#pragma unroll
    for (int mi = 0; mi < 4; ++mi) {
#pragma unroll
        for (int nj = 0; nj < 4; ++nj) {
            const int gn  = nb0 + wn + nj * 16 + lr;
            const float bv = bias[gn];
            const int gm0 = mb0 + wm + mi * 16 + lh * 8;
#pragma unroll
            for (int r = 0; r < 8; ++r) {
                const float y = acc[mi][nj][r] + bv;
                const size_t idx = (size_t)(gm0 + r) * N + gn;
                if (OUT_BF16) Cb[idx] = (__bf16)y;
                else          Cf[idx] = y;
            }
        }
    }
}

// ---------------- per-(b,p) attention: 4 logits, softmax, weighted V sum ----------------
__global__ __launch_bounds__(256) void attn_kernel(const float* __restrict__ Qf,
                                                   const __bf16* __restrict__ Kb,
                                                   const __bf16* __restrict__ Vb,
                                                   float* __restrict__ feat) {
    const int bp  = blockIdx.x;   // 0..4607
    const int tid = threadIdx.x;
    const float*  q = Qf + (size_t)bp * 4096;
    const __bf16* k = Kb + (size_t)bp * 4 * 4096;
    const __bf16* v = Vb + (size_t)bp * 4 * 4096;

    float p[4] = {0.f, 0.f, 0.f, 0.f};
    for (int d = tid * 4; d < 4096; d += 1024) {
        float4 qv = *(const float4*)(q + d);
#pragma unroll
        for (int s = 0; s < 4; ++s) {
            union { uint2 u; __bf16 h[4]; } kk;
            kk.u = *(const uint2*)(k + s * 4096 + d);
            p[s] += qv.x * (float)kk.h[0] + qv.y * (float)kk.h[1] +
                    qv.z * (float)kk.h[2] + qv.w * (float)kk.h[3];
        }
    }
    __shared__ float red[4][256];
#pragma unroll
    for (int s = 0; s < 4; ++s) red[s][tid] = p[s];
    __syncthreads();
    for (int st = 128; st > 0; st >>= 1) {
        if (tid < st) {
#pragma unroll
            for (int s = 0; s < 4; ++s) red[s][tid] += red[s][tid + st];
        }
        __syncthreads();
    }
    float l[4], mx = -1e30f;
#pragma unroll
    for (int s = 0; s < 4; ++s) {
        float t = red[s][0] * 0.015625f;   // 1/sqrt(4096)
        t = (t == t) ? t : 0.f;            // nan_to_num
        l[s] = t; mx = fmaxf(mx, t);
    }
    float e[4], sum = 0.f;
#pragma unroll
    for (int s = 0; s < 4; ++s) { e[s] = __expf(l[s] - mx); sum += e[s]; }
    const float inv = 1.f / sum;
    float wg[4];
#pragma unroll
    for (int s = 0; s < 4; ++s) wg[s] = e[s] * inv;

    for (int d = tid * 4; d < 4096; d += 1024) {
        float a0 = 0.f, a1 = 0.f, a2 = 0.f, a3 = 0.f;
#pragma unroll
        for (int s = 0; s < 4; ++s) {
            union { uint2 u; __bf16 h[4]; } vv;
            vv.u = *(const uint2*)(v + s * 4096 + d);
            a0 += wg[s] * (float)vv.h[0];
            a1 += wg[s] * (float)vv.h[1];
            a2 += wg[s] * (float)vv.h[2];
            a3 += wg[s] * (float)vv.h[3];
        }
        float4 o = {a0, a1, a2, a3};
        *(float4*)(feat + (size_t)bp * 4096 + d) = o;
    }
}

extern "C" void kernel_launch(void* const* d_in, const int* in_sizes, int n_in,
                              void* d_out, int out_size, void* d_ws, size_t ws_size,
                              hipStream_t stream) {
    const float* x_q    = (const float*)d_in[0];
    const float* x_kv   = (const float*)d_in[1];
    const float* q_ln_w = (const float*)d_in[2];
    const float* q_ln_b = (const float*)d_in[3];
    const float* q_w    = (const float*)d_in[4];
    const float* q_b    = (const float*)d_in[5];
    const float* k_ln_w = (const float*)d_in[6];
    const float* k_ln_b = (const float*)d_in[7];
    const float* k_w    = (const float*)d_in[8];
    const float* k_b    = (const float*)d_in[9];
    const float* v_ln_w = (const float*)d_in[10];
    const float* v_ln_b = (const float*)d_in[11];
    const float* v_w    = (const float*)d_in[12];
    const float* v_b    = (const float*)d_in[13];
    float* out = (float*)d_out;

    const size_t N_XQ = (size_t)8 * 576 * 4096;   // 18,874,368 (also feat size)

    // workspace layout (all sizes multiples of 256B)
    char* w = (char*)d_ws;
    __bf16* Aq = (__bf16*)w; w += (size_t)4608  * 4096 * 2;   // LN(x_q)  bf16
    __bf16* Ak = (__bf16*)w; w += (size_t)18432 * 1024 * 2;   // LN_k(kv) bf16 (permuted)
    __bf16* Av = (__bf16*)w; w += (size_t)18432 * 1024 * 2;   // LN_v(kv) bf16 (permuted)
    __bf16* Wq = (__bf16*)w; w += (size_t)4096  * 4096 * 2;
    __bf16* Wk = (__bf16*)w; w += (size_t)1024  * 4096 * 2;
    __bf16* Wv = (__bf16*)w; w += (size_t)1024  * 4096 * 2;
    float*  Qf = (float*) w; w += (size_t)4608  * 4096 * 4;   // q fp32
    __bf16* Kb = (__bf16*)w; w += (size_t)18432 * 4096 * 2;   // k bf16
    __bf16* Vb = (__bf16*)w; w += (size_t)18432 * 4096 * 2;   // v bf16

    // tuple output: (x_q, feat)
    hipMemcpyAsync(out, x_q, N_XQ * sizeof(float), hipMemcpyDeviceToDevice, stream);

    cvt_f32_bf16<<<2048, 256, 0, stream>>>(q_w, Wq, (4096 * 4096) / 4);
    cvt_f32_bf16<<<2048, 256, 0, stream>>>(k_w, Wk, (1024 * 4096) / 4);
    cvt_f32_bf16<<<2048, 256, 0, stream>>>(v_w, Wv, (1024 * 4096) / 4);

    ln_q_kernel<<<4608, 256, 0, stream>>>(x_q, q_ln_w, q_ln_b, Aq);
    ln_kv_kernel<<<18432, 256, 0, stream>>>(x_kv, k_ln_w, k_ln_b, v_ln_w, v_ln_b, Ak, Av);

    gemm_bf16_kernel<false><<<dim3(16, 36),  256, 0, stream>>>(Aq, Wq, q_b, Qf, nullptr, 4608,  4096, 4096);
    gemm_bf16_kernel<true ><<<dim3(16, 144), 256, 0, stream>>>(Ak, Wk, k_b, nullptr, Kb, 18432, 4096, 1024);
    gemm_bf16_kernel<true ><<<dim3(16, 144), 256, 0, stream>>>(Av, Wv, v_b, nullptr, Vb, 18432, 4096, 1024);

    attn_kernel<<<4608, 256, 0, stream>>>(Qf, Kb, Vb, out + N_XQ);
}
